// Llama4DecoderLayer_14740327759922
// MI455X (gfx1250) — matrix operand order repaired and checked by reading, not compiled, not run on hardware
//
#include <hip/hip_runtime.h>

// ---------------------------------------------------------------------------
// Llama4 decoder layer for MI455X (gfx1250, wave32, WMMA bf16 -> f32 accum)
// ---------------------------------------------------------------------------

typedef __bf16 bf16;
typedef bf16  v16bf __attribute__((ext_vector_type(16)));
typedef float v8f   __attribute__((ext_vector_type(8)));

constexpr int   cT   = 2048;
constexpr int   cH   = 2048;
constexpr int   cNH  = 16;
constexpr int   cNKV = 4;
constexpr int   cHD  = 128;
constexpr int   cE   = 8;
constexpr int   cI   = 4096;
constexpr int   cQKV = (cNH + 2 * cNKV) * cHD;    // 3072
constexpr float cEPS = 1e-5f;
constexpr float cSCALE = 0.08838834764831845f;    // 128^-0.5
constexpr float cLOG_THETA = 13.122363377404328f; // ln(500000)

// -------------------------- WMMA helpers -----------------------------------

__device__ inline v8f wmma_bf16(v16bf a, v16bf b, v8f c) {
  return __builtin_amdgcn_wmma_f32_16x16x32_bf16(
      /*neg_a=*/false, a, /*neg_b=*/false, b,
      /*c_mod=*/(short)0, c, /*reuse_a=*/false, /*reuse_b=*/false);
}

// A fragment (16x32, 16-bit): lanes 0-15 hold K{0..7,16..23}, lanes 16-31 hold
// K{8..15,24..31}. Caller passes p = rowbase + (lane>>4)*8; we read p[0..7]
// and p[16..23].
__device__ inline v16bf load_frag_a(const bf16* p) {
  v16bf f;
#pragma unroll
  for (int i = 0; i < 8; i++) { f[i] = p[i]; f[i + 8] = p[i + 16]; }
  return f;
}

// B fragment (32x16, 16-bit) from N-major (transposed) LDS: same K order as
// A. Caller passes p = colbase + (lane>>4)*8; we read p[0..7] and p[16..23].
__device__ inline v16bf load_frag_b(const bf16* p) {
  v16bf f;
#pragma unroll
  for (int i = 0; i < 8; i++) { f[i] = p[i]; f[i + 8] = p[i + 16]; }
  return f;
}

__device__ inline v8f zero8() {
  v8f z;
#pragma unroll
  for (int i = 0; i < 8; i++) z[i] = 0.f;
  return z;
}

// -------------------- fused add + RMSNorm (+bf16 cast) ----------------------
// hs = a + b ; x = rmsnorm(hs, w) -> bf16.  Optionally writes hs (fp32) to
// hs_out and/or res_out.
__global__ __launch_bounds__(256)
void add_rmsnorm_kernel(const float* __restrict__ a, const float* __restrict__ b,
                        const float* __restrict__ w,
                        float* __restrict__ hs_out, float* __restrict__ res_out,
                        bf16* __restrict__ xout) {
  __shared__ float red[256];
  const int t = blockIdx.x, tid = threadIdx.x;
  const long long base = (long long)t * cH;
  float vals[8];
  float ss = 0.f;
#pragma unroll
  for (int i = 0; i < 8; i++) {
    int k = tid + i * 256;
    float v = a[base + k] + b[base + k];
    vals[i] = v;
    ss += v * v;
  }
  red[tid] = ss;
  __syncthreads();
#pragma unroll
  for (int off = 128; off > 0; off >>= 1) {
    if (tid < off) red[tid] += red[tid + off];
    __syncthreads();
  }
  const float inv = rsqrtf(red[0] / (float)cH + cEPS);
#pragma unroll
  for (int i = 0; i < 8; i++) {
    int k = tid + i * 256;
    if (hs_out)  hs_out[base + k]  = vals[i];
    if (res_out) res_out[base + k] = vals[i];
    xout[base + k] = (bf16)(vals[i] * inv * w[k]);
  }
}

// ----------------------------- GEMM ----------------------------------------
// C[M,N] (f32) = gather(A)[M,K] (bf16) x B[K,N] (f32 -> bf16 in LDS)
// Block tile 128x128, K-step 32, 8 waves each own a 32x64 subtile.
// Expert mode (ebase != null): blockIdx.z = expert, rows are the expert's
// token slots; gatherA/scatterC index through perm; accumulate adds into C.
constexpr int GBM = 128, GBN = 128, GBK = 32;
constexpr int LDA_P = 40;  // padded bf16 stride (80B, 16B aligned)
constexpr int LDB_P = 40;

__global__ __launch_bounds__(256)
void gemm_bf16_kernel(const bf16* __restrict__ A, int lda,
                      const float* __restrict__ B, long long strideBe, int ldb,
                      float* __restrict__ C, int ldc,
                      int M, int N, int K,
                      const int* __restrict__ ebase,  // [E+1] or null
                      const int* __restrict__ perm,   // [T] or null
                      int gatherA, int scatterC, int accumulate) {
  __shared__ bf16 As[GBM * LDA_P];
  __shared__ bf16 Bt[GBN * LDB_P];

  const int tid  = threadIdx.x;
  const int lane = tid & 31;
  const int wave = tid >> 5;
  const int half = lane >> 4;
  const int ln   = lane & 15;
  const int n0   = blockIdx.x * GBN;
  const int m0   = blockIdx.y * GBM;

  int rbase = 0, rcount = M;
  if (ebase) { rbase = ebase[blockIdx.z]; rcount = ebase[blockIdx.z + 1] - rbase; }
  if (m0 >= rcount) return;
  const float* Be = B + (long long)blockIdx.z * strideBe;

  const int wm = (wave & 3) * 32;
  const int wn = (wave >> 2) * 64;

  v8f acc[2][4];
#pragma unroll
  for (int i = 0; i < 2; i++)
#pragma unroll
    for (int j = 0; j < 4; j++) acc[i][j] = zero8();

  for (int k0 = 0; k0 < K; k0 += GBK) {
    // ---- stage A: 128x32 bf16 (int4 = 8 bf16 per copy, 2 copies/thread) ----
    {
      int r = tid >> 1;
      int c = (tid & 1) * 16;
      int gr = m0 + r;
      int4 val{0, 0, 0, 0};
      int4 val2{0, 0, 0, 0};
      if (gr < rcount) {
        int gp   = rbase + gr;
        int arow = gatherA ? perm[gp] : gp;
        const int4* src = (const int4*)(A + (long long)arow * lda + k0 + c);
        val  = src[0];
        val2 = src[1];
      }
      int4* dst = (int4*)(As + r * LDA_P + c);
      dst[0] = val;
      dst[1] = val2;
    }
    // ---- stage B transposed: Bt[n][k] = (bf16)B[k0+k][n0+n] ----------------
    {
      int k  = tid >> 3;
      int nb = (tid & 7) * 16;
      const float* src = Be + (long long)(k0 + k) * ldb + n0 + nb;
#pragma unroll
      for (int i = 0; i < 16; i++) Bt[(nb + i) * LDB_P + k] = (bf16)src[i];
    }
    __syncthreads();

    v16bf a0 = load_frag_a(As + (wm + ln) * LDA_P + half * 8);
    v16bf a1 = load_frag_a(As + (wm + 16 + ln) * LDA_P + half * 8);
#pragma unroll
    for (int j = 0; j < 4; j++) {
      v16bf b = load_frag_b(Bt + (wn + j * 16 + ln) * LDB_P + half * 8);
      acc[0][j] = wmma_bf16(a0, b, acc[0][j]);
      acc[1][j] = wmma_bf16(a1, b, acc[1][j]);
    }
    __syncthreads();
  }

  // ---- epilogue: C/D layout row = v + half*8, col = ln ----------------------
#pragma unroll
  for (int i = 0; i < 2; i++) {
#pragma unroll
    for (int j = 0; j < 4; j++) {
#pragma unroll
      for (int v = 0; v < 8; v++) {
        int r  = wm + i * 16 + half * 8 + v;
        int gr = m0 + r;
        if (gr < rcount) {
          int gp   = rbase + gr;
          int crow = scatterC ? perm[gp] : gp;
          float* p = C + (long long)crow * ldc + n0 + wn + j * 16 + ln;
          if (accumulate) *p += acc[i][j][v];
          else            *p  = acc[i][j][v];
        }
      }
    }
  }
}

// ------------------- RoPE + per-head q/k RMS norm --> bf16 ------------------
// blockIdx.x = token, blockIdx.y = head slot (0-15 q, 16-19 k, 20-23 v)
__global__ __launch_bounds__(64)
void rope_norm_kernel(const float* __restrict__ qkv, const int* __restrict__ positions,
                      bf16* __restrict__ qn, bf16* __restrict__ kn,
                      bf16* __restrict__ vn) {
  const int t = blockIdx.x, hh = blockIdx.y, i = threadIdx.x;  // i < 64
  const float* src = qkv + (long long)t * cQKV + hh * cHD;
  if (hh >= cNH + cNKV) {  // v head: plain bf16 cast
    bf16* dst = vn + ((long long)t * cNKV + (hh - cNH - cNKV)) * cHD;
    dst[i]      = (bf16)src[i];
    dst[i + 64] = (bf16)src[i + 64];
    return;
  }
  const int pos = positions[t];
  const float inv_freq = __expf(-cLOG_THETA * (float)i / 64.f);
  const float ang = (float)pos * inv_freq;
  const float c = __cosf(ang), s = __sinf(ang);
  const float x1 = src[i], x2 = src[i + 64];
  const float y1 = x1 * c - x2 * s;
  const float y2 = x2 * c + x1 * s;
  __shared__ float red[64];
  red[i] = y1 * y1 + y2 * y2;
  __syncthreads();
#pragma unroll
  for (int off = 32; off > 0; off >>= 1) {
    if (i < off) red[i] += red[i + off];
    __syncthreads();
  }
  const float inv = rsqrtf(red[0] / (float)cHD + cEPS);
  bf16* dst = (hh < cNH) ? qn + ((long long)t * cNH + hh) * cHD
                         : kn + ((long long)t * cNKV + (hh - cNH)) * cHD;
  dst[i]      = (bf16)(y1 * inv);
  dst[i + 64] = (bf16)(y2 * inv);
}

// ------------------------- flash attention ---------------------------------
// grid (T/128, NH), block 256 (8 waves). Each wave owns a 16-row q strip and
// keeps Q fragments + O accumulators + softmax state in registers; K and V^T
// tiles (64 keys) are staged in LDS; P goes through a per-wave LDS strip to
// become a WMMA-A fragment.
constexpr int FA_LDK = 136;  // K tile row stride (d contiguous), 272B
constexpr int FA_LDV = 72;   // V^T row stride (s contiguous), 144B
constexpr int FA_LDP = 72;

__global__ __launch_bounds__(256)
void flash_attn_kernel(const bf16* __restrict__ Q,   // [T][NH][HD]
                       const bf16* __restrict__ Kk,  // [T][NKV][HD]
                       const bf16* __restrict__ Vv,  // [T][NKV][HD]
                       bf16* __restrict__ O) {       // [T][NH*HD]
  __shared__ bf16 Ks[64 * FA_LDK];
  __shared__ bf16 Vt[128 * FA_LDV];
  __shared__ bf16 Ps[8 * 16 * FA_LDP];

  const int tid = threadIdx.x, lane = tid & 31, wave = tid >> 5;
  const int half = lane >> 4, ln = lane & 15;
  const int head = blockIdx.y, kvh = head >> 2;
  const int qbase = blockIdx.x * 128;

  // Q fragments straight from global (row-major, d contiguous)
  v16bf aq[4];
  {
    const int row = qbase + wave * 16 + ln;
    const bf16* qp = Q + ((long long)row * cNH + head) * cHD + half * 8;
#pragma unroll
    for (int ks = 0; ks < 4; ks++) aq[ks] = load_frag_a(qp + ks * 32);
  }

  v8f o[8];
#pragma unroll
  for (int n = 0; n < 8; n++) o[n] = zero8();
  float mrow[8], lrow[8];
#pragma unroll
  for (int v = 0; v < 8; v++) { mrow[v] = -INFINITY; lrow[v] = 0.f; }

  const int kbmax = 2 * blockIdx.x + 1;
  for (int kb = 0; kb <= kbmax; ++kb) {
    __syncthreads();
    // stage K tile: Ks[s][d] (this *is* B^T for S = Q K^T)
    {
      int s = tid >> 2, d0 = (tid & 3) * 32;
      const bf16* src = Kk + ((long long)(kb * 64 + s) * cNKV + kvh) * cHD + d0;
      int4* dst = (int4*)(Ks + s * FA_LDK + d0);
      const int4* sp = (const int4*)src;
#pragma unroll
      for (int i = 0; i < 4; i++) dst[i] = sp[i];
    }
    // stage V transposed: Vt[d][s]  (B^T for O = P V)
    {
      int s = tid >> 2, d0 = (tid & 3) * 32;
      const bf16* src = Vv + ((long long)(kb * 64 + s) * cNKV + kvh) * cHD + d0;
#pragma unroll
      for (int i = 0; i < 32; i++) Vt[(d0 + i) * FA_LDV + s] = src[i];
    }
    __syncthreads();

    // S = Q K^T (16x64 per wave)
    v8f s4[4];
#pragma unroll
    for (int j = 0; j < 4; j++) s4[j] = zero8();
#pragma unroll
    for (int ks = 0; ks < 4; ks++) {
#pragma unroll
      for (int j = 0; j < 4; j++) {
        v16bf bk = load_frag_b(Ks + (j * 16 + ln) * FA_LDK + ks * 32 + half * 8);
        s4[j] = wmma_bf16(aq[ks], bk, s4[j]);
      }
    }
    // scale + causal mask
    const int rg0 = qbase + wave * 16 + half * 8;
#pragma unroll
    for (int j = 0; j < 4; j++) {
      const int cg = kb * 64 + j * 16 + ln;
#pragma unroll
      for (int v = 0; v < 8; v++) {
        float x = s4[j][v] * cSCALE;
        if (cg > rg0 + v) x = -INFINITY;
        s4[j][v] = x;
      }
    }
    // online softmax (rows private to this wave; reduce across 16-lane half)
    float rmv[8], alpha[8];
#pragma unroll
    for (int v = 0; v < 8; v++) {
      float m = s4[0][v];
#pragma unroll
      for (int j = 1; j < 4; j++) m = fmaxf(m, s4[j][v]);
      rmv[v] = m;
    }
#pragma unroll
    for (int msk = 1; msk <= 8; msk <<= 1)
#pragma unroll
      for (int v = 0; v < 8; v++) rmv[v] = fmaxf(rmv[v], __shfl_xor(rmv[v], msk, 32));
#pragma unroll
    for (int v = 0; v < 8; v++) {
      float mn = fmaxf(mrow[v], rmv[v]);
      alpha[v] = __expf(mrow[v] - mn);
      mrow[v]  = mn;
    }
#pragma unroll
    for (int j = 0; j < 4; j++)
#pragma unroll
      for (int v = 0; v < 8; v++) s4[j][v] = __expf(s4[j][v] - mrow[v]);
    float rsv[8];
#pragma unroll
    for (int v = 0; v < 8; v++) {
      float s = 0.f;
#pragma unroll
      for (int j = 0; j < 4; j++) s += s4[j][v];
      rsv[v] = s;
    }
#pragma unroll
    for (int msk = 1; msk <= 8; msk <<= 1)
#pragma unroll
      for (int v = 0; v < 8; v++) rsv[v] += __shfl_xor(rsv[v], msk, 32);
#pragma unroll
    for (int v = 0; v < 8; v++) lrow[v] = lrow[v] * alpha[v] + rsv[v];
#pragma unroll
    for (int n = 0; n < 8; n++)
#pragma unroll
      for (int v = 0; v < 8; v++) o[n][v] *= alpha[v];

    // P -> per-wave LDS strip -> A fragments (same-wave DS ops are in order)
    bf16* ps = Ps + wave * 16 * FA_LDP;
#pragma unroll
    for (int j = 0; j < 4; j++)
#pragma unroll
      for (int v = 0; v < 8; v++)
        ps[(half * 8 + v) * FA_LDP + j * 16 + ln] = (bf16)s4[j][v];

#pragma unroll
    for (int ks = 0; ks < 2; ks++) {
      v16bf pa = load_frag_a(ps + ln * FA_LDP + ks * 32 + half * 8);
#pragma unroll
      for (int n = 0; n < 8; n++) {
        v16bf bv = load_frag_b(Vt + (n * 16 + ln) * FA_LDV + ks * 32 + half * 8);
        o[n] = wmma_bf16(pa, bv, o[n]);
      }
    }
  }

  // epilogue: normalize and store bf16
#pragma unroll
  for (int n = 0; n < 8; n++) {
#pragma unroll
    for (int v = 0; v < 8; v++) {
      const int row = qbase + wave * 16 + half * 8 + v;
      O[(long long)row * (cNH * cHD) + head * cHD + n * 16 + ln] =
          (bf16)(o[n][v] / lrow[v]);
    }
  }
}

// --------------------------- router (top-1) ---------------------------------
__global__ __launch_bounds__(256)
void router_kernel(const bf16* __restrict__ x2, const float* __restrict__ rw,
                   int* __restrict__ topidx, int* __restrict__ counts,
                   bf16* __restrict__ xw) {
  __shared__ float red[cE][256];
  __shared__ float s_score;
  const int t = blockIdx.x, tid = threadIdx.x;
  const long long base = (long long)t * cH;
  float acc[cE];
#pragma unroll
  for (int e = 0; e < cE; e++) acc[e] = 0.f;
  for (int k = tid; k < cH; k += 256) {
    float xv = (float)x2[base + k];
    const float* r = rw + (long long)k * cE;
#pragma unroll
    for (int e = 0; e < cE; e++) acc[e] += xv * r[e];
  }
#pragma unroll
  for (int e = 0; e < cE; e++) red[e][tid] = acc[e];
  __syncthreads();
  for (int off = 128; off > 0; off >>= 1) {
    if (tid < off)
#pragma unroll
      for (int e = 0; e < cE; e++) red[e][tid] += red[e][tid + off];
    __syncthreads();
  }
  if (tid == 0) {
    float best = red[0][0];
    int bi = 0;
#pragma unroll
    for (int e = 1; e < cE; e++)
      if (red[e][0] > best) { best = red[e][0]; bi = e; }  // first-max tiebreak
    s_score = 1.f / (1.f + __expf(-best));
    topidx[t] = bi;
    atomicAdd(counts + bi, 1);
  }
  __syncthreads();
  const float sc = s_score;
  for (int k = tid; k < cH; k += 256)
    xw[base + k] = (bf16)((float)x2[base + k] * sc);
}

__global__ void prefix_kernel(const int* __restrict__ counts,
                              int* __restrict__ ebase, int* __restrict__ cursor) {
  if (threadIdx.x == 0) {
    int s = 0;
    for (int e = 0; e < cE; e++) { ebase[e] = s; s += counts[e]; cursor[e] = 0; }
    ebase[cE] = s;
  }
}

__global__ __launch_bounds__(256)
void scatter_kernel(const int* __restrict__ topidx, const int* __restrict__ ebase,
                    int* __restrict__ cursor, int* __restrict__ perm) {
  const int t = blockIdx.x * 256 + threadIdx.x;
  if (t < cT) {
    int e = topidx[t];
    int pos = atomicAdd(cursor + e, 1);
    perm[ebase[e] + pos] = t;
  }
}

__global__ __launch_bounds__(256)
void silu_mul_kernel(const float* __restrict__ g, const float* __restrict__ u,
                     bf16* __restrict__ h, long long n) {
  long long i = (long long)blockIdx.x * 256 + threadIdx.x;
  if (i < n) {
    float gv = g[i];
    float s = gv / (1.f + __expf(-gv));
    h[i] = (bf16)(s * u[i]);
  }
}

// ------------------------------ launch --------------------------------------
extern "C" void kernel_launch(void* const* d_in, const int* in_sizes, int n_in,
                              void* d_out, int out_size, void* d_ws, size_t ws_size,
                              hipStream_t stream) {
  const int*   positions = (const int*)d_in[0];
  const float* hidden    = (const float*)d_in[1];
  const float* residual  = (const float*)d_in[2];
  const float* ln1_w     = (const float*)d_in[3];
  const float* ln2_w     = (const float*)d_in[4];
  const float* w_qkv     = (const float*)d_in[5];
  const float* w_o       = (const float*)d_in[6];
  const float* router_w  = (const float*)d_in[7];
  const float* we_gate   = (const float*)d_in[8];
  const float* we_up     = (const float*)d_in[9];
  const float* we_down   = (const float*)d_in[10];
  const float* ws_gate   = (const float*)d_in[11];
  const float* ws_up     = (const float*)d_in[12];
  const float* ws_down   = (const float*)d_in[13];

  float* out  = (float*)d_out;                    // [T*H]
  float* res2 = out + (size_t)cT * cH;            // [T*H]

  // workspace carve-out (~180 MB)
  char* w = (char*)d_ws;
  size_t off = 0;
  auto alloc = [&](size_t bytes) -> char* {
    char* p = w + off;
    off = (off + bytes + 255) & ~(size_t)255;
    return p;
  };
  float* hs    = (float*)alloc((size_t)cT * cH * 4);
  bf16*  x_bf  = (bf16*) alloc((size_t)cT * cH * 2);
  float* qkv   = (float*)alloc((size_t)cT * cQKV * 4);
  bf16*  qn    = (bf16*) alloc((size_t)cT * cNH * cHD * 2);
  bf16*  kn    = (bf16*) alloc((size_t)cT * cNKV * cHD * 2);
  bf16*  vn    = (bf16*) alloc((size_t)cT * cNKV * cHD * 2);
  bf16*  attn  = (bf16*) alloc((size_t)cT * cNH * cHD * 2);
  float* attno = (float*)alloc((size_t)cT * cH * 4);
  bf16*  x2_bf = (bf16*) alloc((size_t)cT * cH * 2);
  bf16*  xw_bf = (bf16*) alloc((size_t)cT * cH * 2);
  float* gbuf  = (float*)alloc((size_t)cT * cI * 4);
  float* ubuf  = (float*)alloc((size_t)cT * cI * 4);
  bf16*  hbuf  = (bf16*) alloc((size_t)cT * cI * 2);
  int*   topidx = (int*)alloc((size_t)cT * 4);
  int*   counts = (int*)alloc((size_t)cE * 4);
  int*   ebase  = (int*)alloc((size_t)(cE + 1) * 4);
  int*   cursor = (int*)alloc((size_t)cE * 4);
  int*   perm   = (int*)alloc((size_t)cT * 4);
  (void)ws_size; (void)in_sizes; (void)n_in; (void)out_size;

  const long long nTI = (long long)cT * cI;

  // 1. hs = hidden + residual ; x = rmsnorm(hs, ln1_w)
  add_rmsnorm_kernel<<<cT, 256, 0, stream>>>(hidden, residual, ln1_w, hs, nullptr, x_bf);
  // 2. qkv = x @ w_qkv
  gemm_bf16_kernel<<<dim3(cQKV / 128, cT / 128, 1), 256, 0, stream>>>(
      x_bf, cH, w_qkv, 0, cQKV, qkv, cQKV, cT, cQKV, cH, nullptr, nullptr, 0, 0, 0);
  // 3. rope + q/k rmsnorm -> bf16
  rope_norm_kernel<<<dim3(cT, cNH + 2 * cNKV), 64, 0, stream>>>(qkv, positions, qn, kn, vn);
  // 4. flash attention
  flash_attn_kernel<<<dim3(cT / 128, cNH), 256, 0, stream>>>(qn, kn, vn, attn);
  // 5. attn @ w_o
  gemm_bf16_kernel<<<dim3(cH / 128, cT / 128, 1), 256, 0, stream>>>(
      attn, cNH * cHD, w_o, 0, cH, attno, cH, cT, cH, cNH * cHD, nullptr, nullptr, 0, 0, 0);
  // 6. hs2 = attno + hs ; res2 -> d_out ; x2 = rmsnorm(hs2, ln2_w)
  add_rmsnorm_kernel<<<cT, 256, 0, stream>>>(attno, hs, ln2_w, nullptr, res2, x2_bf);
  // 7. router top-1 + sigmoid scale
  hipMemsetAsync(counts, 0, cE * sizeof(int), stream);
  router_kernel<<<cT, 256, 0, stream>>>(x2_bf, router_w, topidx, counts, xw_bf);
  prefix_kernel<<<1, 32, 0, stream>>>(counts, ebase, cursor);
  scatter_kernel<<<cT / 256, 256, 0, stream>>>(topidx, ebase, cursor, perm);
  // 8. shared MLP: out = (silu(x2@ws_gate) * (x2@ws_up)) @ ws_down
  gemm_bf16_kernel<<<dim3(cI / 128, cT / 128, 1), 256, 0, stream>>>(
      x2_bf, cH, ws_gate, 0, cI, gbuf, cI, cT, cI, cH, nullptr, nullptr, 0, 0, 0);
  gemm_bf16_kernel<<<dim3(cI / 128, cT / 128, 1), 256, 0, stream>>>(
      x2_bf, cH, ws_up, 0, cI, ubuf, cI, cT, cI, cH, nullptr, nullptr, 0, 0, 0);
  silu_mul_kernel<<<(unsigned)(nTI / 256), 256, 0, stream>>>(gbuf, ubuf, hbuf, nTI);
  gemm_bf16_kernel<<<dim3(cH / 128, cT / 128, 1), 256, 0, stream>>>(
      hbuf, cI, ws_down, 0, cH, out, cH, cT, cH, cI, nullptr, nullptr, 0, 0, 0);
  // 9. routed experts (grouped gather GEMMs), add into out
  gemm_bf16_kernel<<<dim3(cI / 128, cT / 128, cE), 256, 0, stream>>>(
      xw_bf, cH, we_gate, (long long)cH * cI, cI, gbuf, cI, cT, cI, cH,
      ebase, perm, /*gatherA=*/1, /*scatterC=*/0, /*acc=*/0);
  gemm_bf16_kernel<<<dim3(cI / 128, cT / 128, cE), 256, 0, stream>>>(
      xw_bf, cH, we_up, (long long)cH * cI, cI, ubuf, cI, cT, cI, cH,
      ebase, perm, 1, 0, 0);
  silu_mul_kernel<<<(unsigned)(nTI / 256), 256, 0, stream>>>(gbuf, ubuf, hbuf, nTI);
  gemm_bf16_kernel<<<dim3(cH / 128, cT / 128, cE), 256, 0, stream>>>(
      hbuf, cI, we_down, (long long)cI * cH, cH, out, cH, cT, cH, cI,
      ebase, perm, /*gatherA=*/0, /*scatterC=*/1, /*acc=*/1);
}
